// BaseLayerWithLoRA_45260365365949
// MI455X (gfx1250) — matrix operand order repaired and checked by reading, not compiled, not run on hardware
//
#include <hip/hip_runtime.h>

typedef __attribute__((ext_vector_type(16))) __bf16 v16bf;
typedef __attribute__((ext_vector_type(8)))  float  v8f;
typedef __attribute__((ext_vector_type(8)))  __bf16 bf16x8;
typedef __attribute__((ext_vector_type(4)))  __bf16 bf16x4;
typedef __attribute__((ext_vector_type(4)))  float  f32x4;
typedef __attribute__((ext_vector_type(4)))  int    i32x4;

#define DM 8192   // batch * seq
#define DN 4096   // d_out
#define DK 4096   // d_in
#define RR 16     // LoRA rank

#define BM 128
#define BN 128
#define BK 32
#define LSTR 40        // padded LDS row stride in halves (80B: 16B-aligned, bank-conflict-free)
#define BUFH (BM*LSTR) // halves per buffer (A or B)

// gfx1250 async global->LDS path (ASYNCcnt-tracked, no VGPR staging).
#if __has_builtin(__builtin_amdgcn_global_load_async_to_lds_b128) && \
    __has_builtin(__builtin_amdgcn_s_wait_asynccnt)
#define USE_ASYNC 1
// Probed prototype: (int4 as1* src, int4 as3* dst, imm int offset, imm int cpol)
#define ASYNC_CP16(g, l) __builtin_amdgcn_global_load_async_to_lds_b128( \
    (__attribute__((address_space(1))) i32x4*)(g),                       \
    (__attribute__((address_space(3))) i32x4*)(l), 0, 0)
#else
#define USE_ASYNC 0
#endif

__device__ __forceinline__ __bf16 f2bf(float f) {
  union { float f; unsigned u; } in; in.f = f;
  unsigned u = in.u;
  u += 0x7FFFu + ((u >> 16) & 1u);          // round to nearest even
  union { unsigned short s; __bf16 b; } out;
  out.s = (unsigned short)(u >> 16);
  return out.b;
}

// ---------------- Kernel 1: W' = W + B @ A   (f32 -> bf16) ----------------
__global__ __launch_bounds__(256) void fuse_w(const float* __restrict__ W,
                                              const float* __restrict__ A,
                                              const float* __restrict__ B,
                                              __bf16* __restrict__ Wb) {
  const int t = blockIdx.x * 256 + threadIdx.x;  // one thread per 4 elements
  const int o = t >> 10;                         // DK/4 = 1024 quads per row
  const int i = (t & 1023) << 2;
  float b[RR];
#pragma unroll
  for (int r = 0; r < RR; ++r) b[r] = B[o * RR + r];
  f32x4 acc = *(const f32x4*)(W + (size_t)o * DK + i);
#pragma unroll
  for (int r = 0; r < RR; ++r) {
    f32x4 a = *(const f32x4*)(A + (size_t)r * DK + i);
    acc += a * b[r];
  }
  bf16x4 o4;
  o4.x = f2bf(acc.x); o4.y = f2bf(acc.y); o4.z = f2bf(acc.z); o4.w = f2bf(acc.w);
  *(bf16x4*)(Wb + (size_t)o * DK + i) = o4;
}

// ---------------- Kernel 2: x (f32) -> bf16 ----------------
__global__ __launch_bounds__(256) void cvt_x(const float* __restrict__ X,
                                             __bf16* __restrict__ Xb) {
  const size_t t = (size_t)blockIdx.x * 256 + threadIdx.x;
  const size_t base = t * 8;
  f32x4 a = *(const f32x4*)(X + base);
  f32x4 c = *(const f32x4*)(X + base + 4);
  bf16x8 o;
  o[0] = f2bf(a.x); o[1] = f2bf(a.y); o[2] = f2bf(a.z); o[3] = f2bf(a.w);
  o[4] = f2bf(c.x); o[5] = f2bf(c.y); o[6] = f2bf(c.z); o[7] = f2bf(c.w);
  *(bf16x8*)(Xb + base) = o;
}

// ---------------- Kernel 3: out = Xb @ Wb^T + bias  (bf16 WMMA, f32 acc) ----------------
// 256 threads = 8 waves (wave32), tile 128x128, K-step 32, waves 4(M) x 2(N),
// each wave = 32x64 = 2x4 WMMA tiles. Double-buffered async global->LDS staging;
// single-version loop (unroll 1) + pointer toggling so acc stays in fixed VGPRs.
__global__ __launch_bounds__(256) void lora_gemm(const __bf16* __restrict__ Xb,
                                                 const __bf16* __restrict__ Wb,
                                                 const float* __restrict__ bias,
                                                 float* __restrict__ out) {
  __shared__ __align__(16) __bf16 ldsA[2 * BUFH];
  __shared__ __align__(16) __bf16 ldsB[2 * BUFH];

  const int tid  = threadIdx.x;
  const int lane = tid & 31;
  const int wv   = tid >> 5;
  const int wy   = wv & 3;    // wave M index 0..3
  const int wx   = wv >> 2;   // wave N index 0..1
  const int l16  = lane & 15;
  const int hi   = lane >> 4; // 0 = lanes 0-15, 1 = lanes 16-31

  const int m0 = blockIdx.y * BM;
  const int n0 = blockIdx.x * BN;

  // Staging: tile = 128 rows x 32 halves = 512 chunks of 8 halves (16B).
  // Each thread moves chunks (tid) and (tid+256) for both A and B tiles.
  const int rowA0 = tid >> 2,         colA0 = (tid & 3) << 3;
  const int rowA1 = (tid + 256) >> 2, colA1 = ((tid + 256) & 3) << 3;
  const int lo0 = rowA0 * LSTR + colA0;
  const int lo1 = rowA1 * LSTR + colA1;

  // Global source cursors (advance by BK halves per K-step).
  const __bf16* gA0 = Xb + (size_t)(m0 + rowA0) * DK + colA0;
  const __bf16* gA1 = Xb + (size_t)(m0 + rowA1) * DK + colA1;
  const __bf16* gB0 = Wb + (size_t)(n0 + rowA0) * DK + colA0;
  const __bf16* gB1 = Wb + (size_t)(n0 + rowA1) * DK + colA1;

  // LDS stage-destination cursors (toggle between buffers).
  __bf16* wA0 = &ldsA[lo0];
  __bf16* wA1 = &ldsA[lo1];
  __bf16* wB0 = &ldsB[lo0];
  __bf16* wB1 = &ldsB[lo1];

  // Fragment read cursors; all mt/nt/chunk selection folds into imm ds offsets:
  //  A(16x32): lanes 0-15 row=l16 hold K 0..7 + 16..23; lanes 16-31 K 8..15 + 24..31
  //            -> base + {0,16} halves, mt adds 16*LSTR.
  //  B(32x16): col=l16; lanes 0-15 K 0..7 + 16..23; lanes 16-31 K 8..15 + 24..31
  //            -> base + {0,16} halves, nt adds 16*LSTR.
  const __bf16* Af = &ldsA[(wy * 32 + l16) * LSTR + (hi ? 8 : 0)];
  const __bf16* Bf = &ldsB[(wx * 64 + l16) * LSTR + hi * 8];

  v8f acc[2][4] = {};

  auto stage = [&]() {
#if USE_ASYNC
    ASYNC_CP16(gA0, wA0);
    ASYNC_CP16(gA1, wA1);
    ASYNC_CP16(gB0, wB0);
    ASYNC_CP16(gB1, wB1);
#else
    *(bf16x8*)wA0 = *(const bf16x8*)gA0;
    *(bf16x8*)wA1 = *(const bf16x8*)gA1;
    *(bf16x8*)wB0 = *(const bf16x8*)gB0;
    *(bf16x8*)wB1 = *(const bf16x8*)gB1;
#endif
    gA0 += BK; gA1 += BK; gB0 += BK; gB1 += BK;
  };

  auto compute = [&]() {
    union { v16bf v; bf16x8 h[2]; } af[2], bfv[4];
#pragma unroll
    for (int mt = 0; mt < 2; ++mt) {
      af[mt].h[0] = *(const bf16x8*)(Af + mt * 16 * LSTR);
      af[mt].h[1] = *(const bf16x8*)(Af + mt * 16 * LSTR + 16);
    }
#pragma unroll
    for (int nt = 0; nt < 4; ++nt) {
      bfv[nt].h[0] = *(const bf16x8*)(Bf + nt * 16 * LSTR);
      bfv[nt].h[1] = *(const bf16x8*)(Bf + nt * 16 * LSTR + 16);
    }
#pragma unroll
    for (int mt = 0; mt < 2; ++mt)
#pragma unroll
      for (int nt = 0; nt < 4; ++nt)
        acc[mt][nt] = __builtin_amdgcn_wmma_f32_16x16x32_bf16(
            false, af[mt].v, false, bfv[nt].v, (short)0, acc[mt][nt], false, false);
  };

  // Prologue: stage tile 0 into buffer 0; point stage cursors at buffer 1.
  stage();
  int dd = BUFH;
  wA0 += dd; wA1 += dd; wB0 += dd; wB1 += dd;
#if USE_ASYNC
  __builtin_amdgcn_s_wait_asynccnt(0);
#endif
  __syncthreads();

  const int NT = DK / BK;  // 128 K-steps
#pragma unroll 1
  for (int kt = 0; kt < NT - 1; ++kt) {
    stage();     // DMA next tile into the other buffer (overlaps WMMA below)
    compute();   // WMMA on current buffer
#if USE_ASYNC
    __builtin_amdgcn_s_wait_asynccnt(0);
#endif
    __syncthreads();
    // toggle buffers: frags follow the freshly staged tile, stage cursors flip back
    Af += dd; Bf += dd;
    wA0 -= dd; wA1 -= dd; wB0 -= dd; wB1 -= dd;
    dd = -dd;
  }
  compute();     // last tile (already staged)

  // Epilogue: C/D layout — VGPR r, lane L: row = r + 8*(L>=16), col = L%16.
#pragma unroll
  for (int mt = 0; mt < 2; ++mt) {
    const int m = m0 + wy * 32 + mt * 16 + hi * 8;
#pragma unroll
    for (int nt = 0; nt < 4; ++nt) {
      const int n = n0 + wx * 64 + nt * 16 + l16;
      const float bv = bias[n];
      float* p = out + (size_t)m * DN + n;
#pragma unroll
      for (int r = 0; r < 8; ++r)
        p[(size_t)r * DN] = acc[mt][nt][r] + bv;
    }
  }
}

// ---------------- Launch ----------------
extern "C" void kernel_launch(void* const* d_in, const int* in_sizes, int n_in,
                              void* d_out, int out_size, void* d_ws, size_t ws_size,
                              hipStream_t stream) {
  const float* x    = (const float*)d_in[0];
  const float* W    = (const float*)d_in[1];
  const float* bias = (const float*)d_in[2];
  const float* A    = (const float*)d_in[3];
  const float* B    = (const float*)d_in[4];
  float* out = (float*)d_out;

  // Workspace: [0, 32MB) = Wb (bf16 fused weight), [32MB, 96MB) = Xb (bf16 x)
  __bf16* Wb = (__bf16*)d_ws;
  __bf16* Xb = (__bf16*)((char*)d_ws + (size_t)DN * DK * sizeof(unsigned short));

  fuse_w<<<dim3((DN * DK / 4) / 256), dim3(256), 0, stream>>>(W, A, B, Wb);
  cvt_x<<<dim3((DM * DK / 8) / 256), dim3(256), 0, stream>>>(x, Xb);
  lora_gemm<<<dim3(DN / BN, DM / BM), dim3(256), 0, stream>>>(Xb, Wb, bias, out);
}
